// PointTransformerMat_7842610282855
// MI455X (gfx1250) — matrix operand order repaired and checked by reading, not compiled, not run on hardware
//
#include <hip/hip_runtime.h>

typedef __attribute__((ext_vector_type(16))) _Float16 v16h;
typedef __attribute__((ext_vector_type(8)))  _Float16 v8h;
typedef __attribute__((ext_vector_type(8)))  float    v8f;

#define BATCH 4
#define NPTS  8192
#define CH    256
#define NP    128
#define GS    64
#define S_LOC 65
#define SP_LOC 96     // pad to multiple of 32 (K dim of p@v)
#define S_GLB 129
#define SP_GLB 160
#define NWAVE 8

// ---------------------------------------------------------------- WMMA frags
// A 16x32 f16: lane l -> m=l&15 ; halves e0..15 -> k=16*(l>>4)+e.
// Two contiguous 16B LDS loads per lane.
__device__ inline v16h load_a(const _Float16* A, int lda, int m0, int k0, int lane) {
  const _Float16* p = A + (size_t)(m0 + (lane & 15)) * lda + k0 + ((lane >> 4) << 4);
  v8h lo = *(const v8h*)p;
  v8h hi = *(const v8h*)(p + 8);
  v16h r;
#pragma unroll
  for (int i = 0; i < 8; ++i) { r[i] = lo[i]; r[i + 8] = hi[i]; }
  return r;
}
// B 32x16 f16: lane l -> n=l&15 ; k = 16*(l>>4) + e (e=0..15).
// Fragment-packed weights: fragment = 512 contiguous halves, lane-major.
__device__ inline v16h load_b_frag(const _Float16* W, int frag, int lane) {
  return *(const v16h*)(W + ((size_t)frag << 9) + (lane << 4));
}
// B[k][n] = M[n][k] (k^T / vT case): contiguous 32B per lane.
__device__ inline v16h load_b_t(const _Float16* M, int ldm, int k0, int n0, int lane) {
  const _Float16* p = M + (size_t)(n0 + (lane & 15)) * ldm + k0 + ((lane >> 4) << 4);
  return *(const v16h*)p;
}
// C/D f32: lane l -> n=l&15 ; VGPR r -> m = r + 8*(l>>4).
__device__ inline void store_f16_scale(_Float16* O, int ldo, int m0, int n0, int lane,
                                       v8f d, float scale, int Svalid) {
  int n = n0 + (lane & 15);
  int mb = m0 + ((lane >> 4) << 3);
#pragma unroll
  for (int r = 0; r < 8; ++r) {
    int m = mb + r;
    O[(size_t)m * ldo + n] = (m < Svalid) ? (_Float16)(d[r] * scale) : (_Float16)0.f;
  }
}
__device__ inline void store_f16_bias(_Float16* O, int ldo, int m0, int n0, int lane,
                                      v8f d, float bias, int Svalid) {
  int n = n0 + (lane & 15);
  int mb = m0 + ((lane >> 4) << 3);
#pragma unroll
  for (int r = 0; r < 8; ++r) {
    int m = mb + r;
    O[(size_t)m * ldo + n] = (m < Svalid) ? (_Float16)(d[r] + bias) : (_Float16)0.f;
  }
}
// Transposed store (for V): VT[n][m], per-lane contiguous 8 halves in m.
__device__ inline void store_f16_biasT(_Float16* VT, int ldv, int m0, int n0, int lane,
                                       v8f d, float bias, int Svalid) {
  int n = n0 + (lane & 15);
  int mb = m0 + ((lane >> 4) << 3);
#pragma unroll
  for (int r = 0; r < 8; ++r) {
    int m = mb + r;
    VT[(size_t)n * ldv + m] = (m < Svalid) ? (_Float16)(d[r] + bias) : (_Float16)0.f;
  }
}
__device__ inline void accum_res(_Float16* R, int m0, int n0, int lane,
                                 v8f d, float bias, int Svalid) {
  int n = n0 + (lane & 15);
  int mb = m0 + ((lane >> 4) << 3);
#pragma unroll
  for (int r = 0; r < 8; ++r) {
    int m = mb + r;
    if (m < Svalid)
      R[(size_t)m * CH + n] = (_Float16)((float)R[(size_t)m * CH + n] + d[r] + bias);
  }
}

__device__ inline float wave_sum(float v) {
#pragma unroll
  for (int off = 16; off > 0; off >>= 1) v += __shfl_xor(v, off, 32);
  return v;
}
__device__ inline float wave_max(float v) {
#pragma unroll
  for (int off = 16; off > 0; off >>= 1) v = fmaxf(v, __shfl_xor(v, off, 32));
  return v;
}

__device__ inline void ln_block(const _Float16* src, _Float16* dst, const float* g,
                                const float* b, int S, int S_pad, int wv, int lane) {
  for (int t = wv; t < S_pad; t += NWAVE) {
    if (t >= S) {
      for (int c = lane; c < CH; c += 32) dst[(size_t)t * CH + c] = (_Float16)0.f;
      continue;
    }
    float v[8]; float s = 0.f;
#pragma unroll
    for (int i = 0; i < 8; ++i) { v[i] = (float)src[(size_t)t * CH + lane + 32 * i]; s += v[i]; }
    s = wave_sum(s);
    float m = s * (1.f / CH);
    float q = 0.f;
#pragma unroll
    for (int i = 0; i < 8; ++i) { float d = v[i] - m; q += d * d; }
    q = wave_sum(q);
    float rinv = rsqrtf(q * (1.f / CH) + 1e-5f);
#pragma unroll
    for (int i = 0; i < 8; ++i) {
      int c = lane + 32 * i;
      dst[(size_t)t * CH + c] = (_Float16)((v[i] - m) * rinv * g[c] + b[c]);
    }
  }
}

__device__ inline void ln_out(const _Float16* src, float* x, const float* g,
                              const float* b, int S, int wv, int lane) {
  for (int t = wv; t < S; t += NWAVE) {
    float v[8]; float s = 0.f;
#pragma unroll
    for (int i = 0; i < 8; ++i) { v[i] = (float)src[(size_t)t * CH + lane + 32 * i]; s += v[i]; }
    s = wave_sum(s);
    float m = s * (1.f / CH);
    float q = 0.f;
#pragma unroll
    for (int i = 0; i < 8; ++i) { float d = v[i] - m; q += d * d; }
    q = wave_sum(q);
    float rinv = rsqrtf(q * (1.f / CH) + 1e-5f);
#pragma unroll
    for (int i = 0; i < 8; ++i) {
      int c = lane + 32 * i;
      x[(size_t)t * CH + c] = (v[i] - m) * rinv * g[c] + b[c];
    }
  }
}

__device__ inline void softmax_block(_Float16* p, int S, int S_pad, int wv, int lane) {
  for (int t = wv; t < S_pad; t += NWAVE) {
    _Float16* row = p + (size_t)t * S_pad;
    float mx = -3.0e38f;
    for (int c = lane; c < S; c += 32) mx = fmaxf(mx, (float)row[c]);
    mx = wave_max(mx);
    float sm = 0.f;
    for (int c = lane; c < S; c += 32) sm += __expf((float)row[c] - mx);
    sm = wave_sum(sm);
    float inv = 1.f / sm;
    for (int c = lane; c < S_pad; c += 32)
      row[c] = (c < S) ? (_Float16)(__expf((float)row[c] - mx) * inv) : (_Float16)0.f;
  }
}

struct MsaArgs {
  const _Float16* wqkvT[4];   // fragment-packed, K=256, Ncols=768 (48 col tiles)
  const _Float16* woT[4];     // fragment-packed, K=256, Ncols=256 (16 col tiles)
  const _Float16* wfT[4];     // fragment-packed, K=256, Ncols=256
  const float *bqkv[4], *bo[4], *bf[4];
  const float *ln1g[4], *ln1b[4], *ln2g[4], *ln2b[4];
  const float *fng, *fnb;
};

#define WMMA_F16(a, b, c) \
  __builtin_amdgcn_wmma_f32_16x16x32_f16(false, (a), false, (b), (short)0, (c), false, false)

// One block = one sequence. LDS: res[SP][256] f16 | xn[SP][256] f16 |
// qk[SP][128] f16 (q|k per head; o overwrites q) | vT[64][SP] f16 | p[SP][SP] f16.
__global__ __launch_bounds__(256) void msa_kernel(float* xio, int S, int S_pad, MsaArgs P) {
  extern __shared__ _Float16 sm[];
  _Float16* res = sm;                         // [S_pad][256]
  _Float16* xn  = res + (size_t)S_pad * CH;   // [S_pad][256]
  _Float16* qk  = xn + (size_t)S_pad * CH;    // [S_pad][128]
  _Float16* vT  = qk + (size_t)S_pad * 128;   // [64][S_pad]
  _Float16* pbf = vT + (size_t)64 * S_pad;    // [S_pad][S_pad]

  const int tid = threadIdx.x;
  const int lane = tid & 31;
  const int wv = tid >> 5;
  float* x = xio + (size_t)blockIdx.x * S * CH;
  const int Mt = S_pad >> 4;
  const int Ct = S_pad >> 5;

  for (int i = tid; i < S_pad * CH; i += 256) {
    int t = i >> 8;
    res[i] = (t < S) ? (_Float16)x[i] : (_Float16)0.f;
  }
  __syncthreads();

  for (int L = 0; L < 4; ++L) {
    ln_block(res, xn, P.ln1g[L], P.ln1b[L], S, S_pad, wv, lane);
    __syncthreads();

    for (int h = 0; h < 4; ++h) {
      // q|k|v (head h) = xn[S_pad][256] @ WqkvT column tiles
      for (int job = wv; job < Mt * 12; job += NWAVE) {
        int mt = job % Mt;
        int pn = job / Mt;
        int part = pn >> 2, nt = pn & 3;         // part: 0=q 1=k 2=v
        int ct = part * 16 + h * 4 + nt;         // column tile among 48
        v8f acc = {};
#pragma unroll
        for (int kk = 0; kk < 8; ++kk) {
          v16h a = load_a(xn, CH, mt * 16, kk * 32, lane);
          v16h b = load_b_frag(P.wqkvT[L], kk * 48 + ct, lane);
          acc = WMMA_F16(a, b, acc);
        }
        float bias = P.bqkv[L][ct * 16 + (lane & 15)];
        if (part < 2)
          store_f16_bias(qk, 128, mt * 16, part * 64 + nt * 16, lane, acc, bias, S);
        else
          store_f16_biasT(vT, S_pad, mt * 16, nt * 16, lane, acc, bias, S);
      }
      __syncthreads();
      // scores = q @ k^T * 1/sqrt(64)
      for (int job = wv; job < Mt * Mt; job += NWAVE) {
        int mt = job % Mt, nt = job / Mt;
        v8f acc = {};
#pragma unroll
        for (int kk = 0; kk < 2; ++kk) {
          v16h a = load_a(qk, 128, mt * 16, kk * 32, lane);
          v16h b = load_b_t(qk + 64, 128, kk * 32, nt * 16, lane);
          acc = WMMA_F16(a, b, acc);
        }
        store_f16_scale(pbf, S_pad, mt * 16, nt * 16, lane, acc, 0.125f, S_pad);
      }
      __syncthreads();
      softmax_block(pbf, S, S_pad, wv, lane);
      __syncthreads();
      // o_h = p @ v  (B[k][n] = vT[n][k]) -> overwrite q slice (cols 0..63)
      for (int job = wv; job < Mt * 4; job += NWAVE) {
        int mt = job % Mt, nt = job / Mt;
        v8f acc = {};
        for (int kk = 0; kk < Ct; ++kk) {
          v16h a = load_a(pbf, S_pad, mt * 16, kk * 32, lane);
          v16h b = load_b_t(vT, S_pad, kk * 32, nt * 16, lane);
          acc = WMMA_F16(a, b, acc);
        }
        store_f16_scale(qk, 128, mt * 16, nt * 16, lane, acc, 1.f, S_pad);
      }
      __syncthreads();
      // res += o_h @ WoT[h*64 : h*64+64][:]  (+bo once)
      for (int job = wv; job < Mt * 16; job += NWAVE) {
        int mt = job % Mt, nt = job / Mt;
        v8f acc = {};
#pragma unroll
        for (int kk = 0; kk < 2; ++kk) {
          v16h a = load_a(qk, 128, mt * 16, kk * 32, lane);
          v16h b = load_b_frag(P.woT[L], (h * 2 + kk) * 16 + nt, lane);
          acc = WMMA_F16(a, b, acc);
        }
        float bias = (h == 0) ? P.bo[L][nt * 16 + (lane & 15)] : 0.f;
        accum_res(res, mt * 16, nt * 16, lane, acc, bias, S);
      }
      __syncthreads();
    }

    ln_block(res, xn, P.ln2g[L], P.ln2b[L], S, S_pad, wv, lane);
    __syncthreads();
    // res += xn @ WfT + bf
    for (int job = wv; job < Mt * 16; job += NWAVE) {
      int mt = job % Mt, nt = job / Mt;
      v8f acc = {};
#pragma unroll
      for (int kk = 0; kk < 8; ++kk) {
        v16h a = load_a(xn, CH, mt * 16, kk * 32, lane);
        v16h b = load_b_frag(P.wfT[L], kk * 16 + nt, lane);
        acc = WMMA_F16(a, b, acc);
      }
      float bias = P.bf[L][nt * 16 + (lane & 15)];
      accum_res(res, mt * 16, nt * 16, lane, acc, bias, S);
    }
    __syncthreads();
  }
  ln_out(res, x, P.fng, P.fnb, S, wv, lane);
}

// ---------------------------------------------------------------- aux kernels
// Pack f32 weight [Nc][K] (row-major, [out][in]) into WMMA B-fragment order:
// fragment (kt,nt) = 512 halves; half i: lane=(i>>4)&31, e=i&15,
// k = kt*32 + 16*(lane>>4) + e, n = nt*16 + (lane&15).
__global__ void pack_w_kernel(const float* __restrict__ src, _Float16* __restrict__ dst,
                              int K, int Nc) {
  int total = K * Nc;
  int nt16 = Nc >> 4;
  for (int i = blockIdx.x * blockDim.x + threadIdx.x; i < total; i += gridDim.x * blockDim.x) {
    int frag = i >> 9, slot = i & 511;
    int lane = slot >> 4, e = slot & 15;
    int kt = frag / nt16, nt = frag % nt16;
    int k = kt * 32 + ((lane >> 4) << 4) + e;
    int n = nt * 16 + (lane & 15);
    dst[i] = (_Float16)src[(size_t)n * K + k];
  }
}

__global__ __launch_bounds__(256) void fps_kernel(const float* __restrict__ xyz,
                                                  int* __restrict__ fps_idx) {
  __shared__ float dmin[NPTS];
  __shared__ float rv[256];
  __shared__ int ri[256];
  __shared__ int s_last;
  const int b = blockIdx.x, tid = threadIdx.x;
  const float* p = xyz + (size_t)b * NPTS * 3;
  for (int n = tid; n < NPTS; n += 256) dmin[n] = 1e10f;
  if (tid == 0) { fps_idx[b * NP] = 0; s_last = 0; }
  __syncthreads();
  for (int it = 1; it < NP; ++it) {
    int last = s_last;
    float lx = p[last * 3], ly = p[last * 3 + 1], lz = p[last * 3 + 2];
    float bv = -1.f; int bi = 0;
    for (int n = tid; n < NPTS; n += 256) {
      float dx = p[n * 3] - lx, dy = p[n * 3 + 1] - ly, dz = p[n * 3 + 2] - lz;
      float dm = fminf(dmin[n], dx * dx + dy * dy + dz * dz);
      dmin[n] = dm;
      if (dm > bv || (dm == bv && n < bi)) { bv = dm; bi = n; }
    }
    rv[tid] = bv; ri[tid] = bi;
    __syncthreads();
    for (int s = 128; s > 0; s >>= 1) {
      if (tid < s) {
        float ov = rv[tid + s]; int oi = ri[tid + s];
        if (ov > rv[tid] || (ov == rv[tid] && oi < ri[tid])) { rv[tid] = ov; ri[tid] = oi; }
      }
      __syncthreads();
    }
    if (tid == 0) { s_last = ri[0]; fps_idx[b * NP + it] = ri[0]; }
    __syncthreads();
  }
}

__global__ __launch_bounds__(256) void knn_kernel(const float* __restrict__ xyz,
                                                  const int* __restrict__ fps_idx,
                                                  int* __restrict__ knn_idx) {
  __shared__ float d2[NPTS];
  __shared__ float rv[256];
  __shared__ int ri[256];
  int bj = blockIdx.x, b = bj >> 7, tid = threadIdx.x;
  const float* p = xyz + (size_t)b * NPTS * 3;
  int ci = fps_idx[bj];
  float cx = p[ci * 3], cy = p[ci * 3 + 1], cz = p[ci * 3 + 2];
  for (int n = tid; n < NPTS; n += 256) {
    float dx = p[n * 3] - cx, dy = p[n * 3 + 1] - cy, dz = p[n * 3 + 2] - cz;
    d2[n] = dx * dx + dy * dy + dz * dz;
  }
  __syncthreads();
  for (int k = 0; k < GS; ++k) {
    float bv = 3.0e38f; int bi = 0;
    for (int n = tid; n < NPTS; n += 256) {
      float v = d2[n];
      if (v < bv || (v == bv && n < bi)) { bv = v; bi = n; }
    }
    rv[tid] = bv; ri[tid] = bi;
    __syncthreads();
    for (int s = 128; s > 0; s >>= 1) {
      if (tid < s) {
        float ov = rv[tid + s]; int oi = ri[tid + s];
        if (ov < rv[tid] || (ov == rv[tid] && oi < ri[tid])) { rv[tid] = ov; ri[tid] = oi; }
      }
      __syncthreads();
    }
    if (tid == 0) { knn_idx[bj * GS + k] = ri[0]; d2[ri[0]] = 3.4e38f; }
    __syncthreads();
  }
}

__global__ __launch_bounds__(256) void group_kernel(const float* __restrict__ pd,
                                                    const int* __restrict__ knn,
                                                    float* __restrict__ seq,
                                                    float* __restrict__ cent) {
  int bj = blockIdx.x, b = bj >> 7, c = threadIdx.x;
  for (int t = 0; t < GS; ++t) {
    int src = knn[bj * GS + t];
    seq[((size_t)bj * S_LOC + t) * CH + c] = pd[((size_t)b * NPTS + src) * CH + c];
  }
  seq[((size_t)bj * S_LOC + GS) * CH + c] = 0.f;  // cluster token
  if (c < 3) {
    float s = 0.f;
    for (int t = 0; t < GS; ++t) s += pd[((size_t)b * NPTS + knn[bj * GS + t]) * CH + c];
    cent[bj * 3 + c] = s * (1.f / GS);
  }
}

__global__ void closest_kernel(const float* __restrict__ xyz, const float* __restrict__ cent,
                               int* __restrict__ closest) {
  int i = blockIdx.x * blockDim.x + threadIdx.x;
  if (i >= BATCH * NPTS) return;
  int b = i / NPTS;
  float x = xyz[(size_t)i * 3], y = xyz[(size_t)i * 3 + 1], z = xyz[(size_t)i * 3 + 2];
  float bv = 3.4e38f; int bi = 0;
  for (int j = 0; j < NP; ++j) {
    float dx = x - cent[(b * NP + j) * 3];
    float dy = y - cent[(b * NP + j) * 3 + 1];
    float dz = z - cent[(b * NP + j) * 3 + 2];
    float d = dx * dx + dy * dy + dz * dz;
    if (d < bv) { bv = d; bi = j; }
  }
  closest[i] = bi;
}

__global__ void zero_kernel(float* p, size_t n) {
  for (size_t i = blockIdx.x * (size_t)blockDim.x + threadIdx.x; i < n;
       i += (size_t)gridDim.x * blockDim.x)
    p[i] = 0.f;
}

__global__ __launch_bounds__(256) void scatter_kernel(const float* __restrict__ seq,
                                                      const int* __restrict__ knn,
                                                      float* __restrict__ rec,
                                                      float* __restrict__ cnt) {
  int bj = blockIdx.x, b = bj >> 7, c = threadIdx.x;
  for (int t = 0; t < GS; ++t) {
    int dst = knn[bj * GS + t];
    atomicAdd(&rec[((size_t)b * NPTS + dst) * CH + c], seq[((size_t)bj * S_LOC + t) * CH + c]);
  }
  if (c < GS) atomicAdd(&cnt[(size_t)b * NPTS + knn[bj * GS + c]], 1.f);
}

__global__ void cf_kernel(const float* __restrict__ seq, float* __restrict__ cf) {
  int i = blockIdx.x * blockDim.x + threadIdx.x;
  if (i >= BATCH * S_GLB * CH) return;
  int c = i % CH, j = (i / CH) % S_GLB, b = i / (CH * S_GLB);
  cf[i] = (j < NP) ? seq[(((size_t)(b * NP + j)) * S_LOC + GS) * CH + c] : 0.f;
}

__global__ void final_kernel(const float* __restrict__ pd, const float* __restrict__ rec,
                             const float* __restrict__ cnt, const float* __restrict__ cf,
                             const int* __restrict__ closest, float* __restrict__ out) {
  size_t total = (size_t)BATCH * NPTS * CH;
  for (size_t i = blockIdx.x * (size_t)blockDim.x + threadIdx.x; i < total;
       i += (size_t)gridDim.x * blockDim.x) {
    size_t c = i % CH, bn = i / CH, b = bn / NPTS;
    out[bn * 512 + c] = pd[i] + rec[i] / fmaxf(cnt[bn], 1.f);
    out[bn * 512 + CH + c] = cf[((size_t)b * S_GLB + closest[bn]) * CH + c];
  }
  size_t gi = blockIdx.x * (size_t)blockDim.x + threadIdx.x;
  if (gi < (size_t)BATCH * CH) {
    size_t b = gi / CH, c = gi % CH;
    out[(size_t)BATCH * NPTS * 512 + gi] = cf[((size_t)b * S_GLB + NP) * CH + c];
  }
}

// ---------------------------------------------------------------------- host
extern "C" void kernel_launch(void* const* d_in, const int* in_sizes, int n_in,
                              void* d_out, int out_size, void* d_ws, size_t ws_size,
                              hipStream_t stream) {
  (void)in_sizes; (void)n_in; (void)out_size; (void)ws_size;
  const float* xyz = (const float*)d_in[0];
  const float* pd  = (const float*)d_in[1];
  // pytree order: 'global' at 2, 'local' at 44; per-set: fn_b, fn_g, 4x layer
  // layer keys: bf,bo,bqkv,ln1_b,ln1_g,ln2_b,ln2_g,wf,wo,wqkv
  const int GBASE = 2, LBASE = 44;
  auto lp = [&](int base, int L, int w) { return (const float*)d_in[base + 2 + L * 10 + w]; };

  size_t off = 0;
  auto carve = [&](size_t bytes) -> void* {
    void* p = (char*)d_ws + off;
    off += (bytes + 255) & ~(size_t)255;
    return p;
  };
  float* rec = (float*)carve((size_t)BATCH * NPTS * CH * 4);
  float* cnt = (float*)carve((size_t)BATCH * NPTS * 4);
  float* seq = (float*)carve((size_t)BATCH * NP * S_LOC * CH * 4);
  float* cf  = (float*)carve((size_t)BATCH * S_GLB * CH * 4);
  int* fpsi  = (int*)carve((size_t)BATCH * NP * 4);
  int* knn   = (int*)carve((size_t)BATCH * NP * GS * 4);
  float* cent = (float*)carve((size_t)BATCH * NP * 3 * 4);
  int* clo   = (int*)carve((size_t)BATCH * NPTS * 4);
  const size_t PLH = 196608 + 65536 + 65536;  // per-layer f16 weight elems
  _Float16* wbuf = (_Float16*)carve((size_t)2 * 4 * PLH * 2);

  int bases[2] = {LBASE, GBASE};  // set 0 = local, set 1 = global
  for (int s = 0; s < 2; ++s)
    for (int L = 0; L < 4; ++L) {
      _Float16* dl = wbuf + ((size_t)s * 4 + L) * PLH;
      pack_w_kernel<<<768, 256, 0, stream>>>(lp(bases[s], L, 9), dl, 256, 768);            // Wqkv
      pack_w_kernel<<<256, 256, 0, stream>>>(lp(bases[s], L, 8), dl + 196608, 256, 256);   // Wo
      pack_w_kernel<<<256, 256, 0, stream>>>(lp(bases[s], L, 7), dl + 262144, 256, 256);   // Wf
    }

  fps_kernel<<<BATCH, 256, 0, stream>>>(xyz, fpsi);
  knn_kernel<<<BATCH * NP, 256, 0, stream>>>(xyz, fpsi, knn);
  group_kernel<<<BATCH * NP, 256, 0, stream>>>(pd, knn, seq, cent);
  closest_kernel<<<(BATCH * NPTS + 255) / 256, 256, 0, stream>>>(xyz, cent, clo);

  MsaArgs PA[2];
  for (int s = 0; s < 2; ++s) {
    int base = bases[s];
    PA[s].fnb = (const float*)d_in[base + 0];
    PA[s].fng = (const float*)d_in[base + 1];
    for (int L = 0; L < 4; ++L) {
      _Float16* dl = wbuf + ((size_t)s * 4 + L) * PLH;
      PA[s].wqkvT[L] = dl;
      PA[s].woT[L] = dl + 196608;
      PA[s].wfT[L] = dl + 262144;
      PA[s].bf[L] = lp(base, L, 0);
      PA[s].bo[L] = lp(base, L, 1);
      PA[s].bqkv[L] = lp(base, L, 2);
      PA[s].ln1b[L] = lp(base, L, 3);
      PA[s].ln1g[L] = lp(base, L, 4);
      PA[s].ln2b[L] = lp(base, L, 5);
      PA[s].ln2g[L] = lp(base, L, 6);
    }
  }
  size_t lds_loc = (size_t)(SP_LOC * (CH * 2 + 128 + 64) + SP_LOC * SP_LOC) * 2;  // 150 KB
  size_t lds_glb = (size_t)(SP_GLB * (CH * 2 + 128 + 64) + SP_GLB * SP_GLB) * 2;  // 270 KB
  hipFuncSetAttribute((const void*)msa_kernel, hipFuncAttributeMaxDynamicSharedMemorySize,
                      (int)lds_glb);

  msa_kernel<<<BATCH * NP, 256, lds_loc, stream>>>(seq, S_LOC, SP_LOC, PA[0]);

  zero_kernel<<<2048, 256, 0, stream>>>(rec, (size_t)BATCH * NPTS * CH);
  zero_kernel<<<64, 256, 0, stream>>>(cnt, (size_t)BATCH * NPTS);
  scatter_kernel<<<BATCH * NP, 256, 0, stream>>>(seq, knn, rec, cnt);
  cf_kernel<<<(BATCH * S_GLB * CH + 255) / 256, 256, 0, stream>>>(seq, cf);

  msa_kernel<<<BATCH, 256, lds_glb, stream>>>(cf, S_GLB, SP_GLB, PA[1]);

  final_kernel<<<4096, 256, 0, stream>>>(pd, rec, cnt, cf, clo, (float*)d_out);
}
